// InvariantMultiACE_88390426951950
// MI455X (gfx1250) — compile-verified, matrix operand order checked by reading, not run on hardware
//
#include <hip/hip_runtime.h>
#include <hip/hip_bf16.h>

// ---------------- problem constants (from reference) ----------------
static constexpr int   Nn    = 8192;
static constexpr int   Ne    = 65536;
static constexpr int   Gg    = 16;
static constexpr int   NELEM = 10;
static constexpr int   Cc    = 64;
static constexpr int   NBas  = 8;
static constexpr int   LMc   = 16;          // (LMAX+1)^2
static constexpr int   Ff    = 256;         // (LMAX+1)*C
static constexpr float R_MAX = 5.0f;
static constexpr float EPSf  = 1e-9f;

typedef _Float16 v16h __attribute__((ext_vector_type(16)));
typedef _Float16 v8h  __attribute__((ext_vector_type(8)));
typedef float    v8f  __attribute__((ext_vector_type(8)));
typedef int      v4i  __attribute__((ext_vector_type(4)));
typedef __attribute__((address_space(1))) v4i* gp_v4i;   // global int4*
typedef __attribute__((address_space(3))) v4i* sp_v4i;   // LDS int4*

// gfx1250 async global->LDS path (probe-guarded; falls back to plain staging)
#if defined(__has_builtin)
# if __has_builtin(__builtin_amdgcn_global_load_async_to_lds_b128) && \
     __has_builtin(__builtin_amdgcn_s_wait_asynccnt)
#  define HAVE_ASYNC_LDS 1
# endif
#endif
#ifndef HAVE_ASYNC_LDS
# define HAVE_ASYNC_LDS 0
#endif

// ---------------- small helpers ----------------
__device__ __forceinline__ int lblock(int lm) {
  return (lm >= 9) ? 3 : (lm >= 4) ? 2 : (lm >= 1) ? 1 : 0;
}

// ---------------- elem = argmax(one-hot) ----------------
__global__ void k_elem(const float* __restrict__ attrs, int* __restrict__ elem) {
  int n = blockIdx.x * blockDim.x + threadIdx.x;
  if (n >= Nn) return;
  int best = 0; float bv = attrs[n * NELEM];
  for (int e = 1; e < NELEM; ++e) {
    float v = attrs[n * NELEM + e];
    if (v > bv) { bv = v; best = e; }
  }
  elem[n] = best;
}

// ---------------- h0 = W_embed[elem] ----------------
__global__ void k_embed(const int* __restrict__ elem, const float* __restrict__ Wemb,
                        float* __restrict__ h0) {
  int t = blockIdx.x * blockDim.x + threadIdx.x;
  if (t >= Nn * Cc) return;
  int n = t >> 6, c = t & 63;
  h0[t] = Wemb[elem[n] * Cc + c];
}

// ---------------- f32 -> f16 streaming convert ----------------
__global__ void k_cvt(const float* __restrict__ src, _Float16* __restrict__ dst, int n) {
  int t = blockIdx.x * blockDim.x + threadIdx.x;
  if (t < n) dst[t] = (_Float16)src[t];
}

// ---------------- per-edge geometry: u, r, Y[16], Rb[8], dRdr[8] ----------------
__global__ void k_geom(const float* __restrict__ pos, const float* __restrict__ shifts,
                       const int* __restrict__ eidx,
                       float* __restrict__ uu, float* __restrict__ rr,
                       float* __restrict__ Y, float* __restrict__ Rb,
                       float* __restrict__ dRdr) {
  int e = blockIdx.x * blockDim.x + threadIdx.x;
  if (e >= Ne) return;
  int snd = eidx[e], rcv = eidx[Ne + e];
  float vx = pos[rcv * 3 + 0] - pos[snd * 3 + 0] + shifts[e * 3 + 0];
  float vy = pos[rcv * 3 + 1] - pos[snd * 3 + 1] + shifts[e * 3 + 1];
  float vz = pos[rcv * 3 + 2] - pos[snd * 3 + 2] + shifts[e * 3 + 2];
  float r  = sqrtf(vx * vx + vy * vy + vz * vz + EPSf);
  float inv = 1.0f / r;
  float x = vx * inv, y = vy * inv, z = vz * inv;
  uu[e * 3 + 0] = x; uu[e * 3 + 1] = y; uu[e * 3 + 2] = z;
  rr[e] = r;
  float x2 = x * x, y2 = y * y, z2 = z * z;
  const float c0 = 0.28209479177387814f, c1 = 0.4886025119029199f;
  const float c4 = 1.0925484305920792f, c6 = 0.31539156525252005f;
  const float c8 = 0.5462742152960396f, c9 = 0.5900435899266435f;
  const float c10 = 2.890611442640554f, c11 = 0.4570457994644658f;
  const float c12 = 0.3731763325901154f, c14 = 1.445305721320277f;
  float* Ye = Y + (size_t)e * LMc;
  Ye[0]  = c0;
  Ye[1]  = c1 * y;  Ye[2]  = c1 * z;  Ye[3]  = c1 * x;
  Ye[4]  = c4 * x * y; Ye[5] = c4 * y * z;
  Ye[6]  = c6 * (3.0f * z2 - 1.0f);
  Ye[7]  = c4 * x * z; Ye[8] = c8 * (x2 - y2);
  Ye[9]  = c9 * y * (3.0f * x2 - y2);
  Ye[10] = c10 * x * y * z;
  Ye[11] = c11 * y * (5.0f * z2 - 1.0f);
  Ye[12] = c12 * z * (5.0f * z2 - 3.0f);
  Ye[13] = c11 * x * (5.0f * z2 - 1.0f);
  Ye[14] = c14 * z * (x2 - y2);
  Ye[15] = c9 * x * (x2 - 3.0f * y2);
  // radial
  float u = r / R_MAX;
  float cut = 0.0f, dcut = 0.0f;
  if (u < 1.0f) {
    float u2 = u * u, u4 = u2 * u2, u5 = u4 * u, u6 = u5 * u, u7 = u6 * u;
    cut  = 1.0f - 21.0f * u5 + 35.0f * u6 - 15.0f * u7;
    dcut = (-105.0f * u4 + 210.0f * u5 - 105.0f * u6) / R_MAX;
  }
  const float SQ = 0.6324555320336759f; // sqrt(2/R_MAX)
  const float PI = 3.14159265358979323846f;
  for (int b = 0; b < NBas; ++b) {
    float np  = (float)(b + 1) * PI / R_MAX;
    float s = __sinf(np * r), co = __cosf(np * r);
    float bes  = SQ * s * inv;
    float dbes = SQ * (np * co * inv - s * inv * inv);
    Rb[(size_t)e * NBas + b]   = bes * cut;
    dRdr[(size_t)e * NBas + b] = dbes * cut + bes * dcut;
  }
}

// ---------------- wgt[e,c] = sum_b Rb[e,b] * W_rad[b,c] ----------------
__global__ void k_radW(const float* __restrict__ Rb, const float* __restrict__ Wr,
                       float* __restrict__ wgt) {
  int t = blockIdx.x * blockDim.x + threadIdx.x;
  if (t >= Ne * Cc) return;
  int e = t >> 6, c = t & 63;
  float acc = 0.0f;
  for (int b = 0; b < NBas; ++b) acc += Rb[(size_t)e * NBas + b] * Wr[b * Cc + c];
  wgt[t] = acc;
}

// ---------------- scatter: A[rcv,lm,c] += (wgt*h[snd])_c * Y_lm ----------------
__global__ void k_scatter(const int* __restrict__ eidx, const float* __restrict__ wgt,
                          const float* __restrict__ h, const float* __restrict__ Y,
                          float* __restrict__ A) {
  __shared__ float sY[LMc];
  int e = blockIdx.x, c = threadIdx.x;
  if (c < LMc) sY[c] = Y[(size_t)e * LMc + c];
  __syncthreads();
  int snd = eidx[e], rcv = eidx[Ne + e];
  float wh = wgt[(size_t)e * Cc + c] * h[(size_t)snd * Cc + c];
  float* Ar = A + (size_t)rcv * (LMc * Cc) + c;
  #pragma unroll
  for (int lm = 0; lm < LMc; ++lm)
    atomicAdd(Ar + lm * Cc, wh * sY[lm]);
}

// ---------------- B[n, l*C+c] = sum_{lm in block l} A^2 ----------------
__global__ void k_poolB(const float* __restrict__ A, float* __restrict__ B) {
  int t = blockIdx.x * blockDim.x + threadIdx.x;
  if (t >= Nn * Cc) return;
  int n = t >> 6, c = t & 63;
  const float* Ar = A + (size_t)n * (LMc * Cc) + c;
  float s0 = 0, s1 = 0, s2 = 0, s3 = 0;
  #pragma unroll
  for (int lm = 0; lm < LMc; ++lm) {
    float v = Ar[lm * Cc]; v *= v;
    int l = lblock(lm);
    s0 += (l == 0) ? v : 0.0f; s1 += (l == 1) ? v : 0.0f;
    s2 += (l == 2) ? v : 0.0f; s3 += (l == 3) ? v : 0.0f;
  }
  float* Bn = B + (size_t)n * Ff;
  Bn[c] = s0; Bn[Cc + c] = s1; Bn[2 * Cc + c] = s2; Bn[3 * Cc + c] = s3;
}

// ---------------- pack W (f32 [NELEM,K,ND]) into WMMA B-fragment layout (f16) ----
// Wp[((el*NT + nt)*NK + ks)*32 + lane][16 halves]
template <int K, int ND>
__global__ void k_packW(const float* __restrict__ Wf, _Float16* __restrict__ Wp) {
  constexpr int NK = K / 32, NT = ND / 16;
  int t = blockIdx.x * blockDim.x + threadIdx.x;
  if (t >= NELEM * NT * NK * 32) return;
  int lane = t & 31, rest = t >> 5;
  int ks = rest % NK; rest /= NK;
  int nt = rest % NT; int el = rest / NT;
  int r16 = lane & 15, kb = (lane >> 4) * 8;
  int ncol = nt * 16 + r16;
  const float* src = Wf + (size_t)el * K * ND + ncol;
  _Float16* dst = Wp + (size_t)t * 16;
  #pragma unroll
  for (int j = 0; j < 8; ++j) {
    dst[j]     = (_Float16)src[(size_t)(ks * 32 + kb + j) * ND];
    dst[j + 8] = (_Float16)src[(size_t)(ks * 32 + 16 + kb + j) * ND];
  }
}

// ---------------- WMMA GEMM, 4 waves/block, LDS-staged pre-packed weights -------
// D[m, 0:ND] = Ah[m, 0:K] @ W[elem]; rows stored only where elem[m]==blockIdx.z
template <int K, int ND>
__global__ void k_gemm2(const _Float16* __restrict__ Ah, const _Float16* __restrict__ Wp,
                        const int* __restrict__ elem, float* __restrict__ D) {
  constexpr int NK = K / 32;          // k-steps of 32
  constexpr int NT = ND / 16;         // total 16-col tiles
  constexpr int HALVES = 4 * NK * 32 * 16;   // staged halves (4 tiles per block)
  __shared__ __align__(16) _Float16 sW[HALVES];

  const int m0 = blockIdx.x * 16;
  const int el = blockIdx.z;
  const int tid = threadIdx.x;

  // block-uniform early exit (same 16 rows for every thread -> EXEC stays full)
  bool found = false;
  #pragma unroll
  for (int j = 0; j < 16; ++j) found |= (elem[m0 + j] == el);
  if (!found) return;

  const int lane = tid & 31, w = tid >> 5;
  const int r16 = lane & 15, kb = (lane >> 4) * 8;
  const int row = m0 + r16;
  __builtin_prefetch(Ah + (size_t)row * K, 0, 1);   // global_prefetch_b8

  const int ntg0 = blockIdx.y * 4;
  const _Float16* gsrc = Wp + ((size_t)el * NT + ntg0) * NK * 32 * 16;
  constexpr int CHUNKS = HALVES * 2 / 16;           // 16-byte chunks
#if HAVE_ASYNC_LDS
  for (int i = tid; i < CHUNKS; i += 128) {
    __builtin_amdgcn_global_load_async_to_lds_b128(
        (gp_v4i)(gsrc + (size_t)i * 8),
        (sp_v4i)(sW + (size_t)i * 8), 0, 0);
  }
  __builtin_amdgcn_s_wait_asynccnt(0);
#else
  for (int i = tid; i < CHUNKS; i += 128)
    ((uint4*)sW)[i] = ((const uint4*)gsrc)[i];
#endif
  __syncthreads();

  // A fragments: two contiguous 16B vector loads per k-step
  v16h a[NK];
  const _Float16* Arp = Ah + (size_t)row * K + kb;
  #pragma unroll
  for (int ks = 0; ks < NK; ++ks) {
    v8h lo = *(const v8h*)(Arp + ks * 32);
    v8h hi = *(const v8h*)(Arp + ks * 32 + 16);
    v16h t;
    #pragma unroll
    for (int j = 0; j < 8; ++j) { t[j] = lo[j]; t[j + 8] = hi[j]; }
    a[ks] = t;
  }

  v8f acc = {};
  const _Float16* sWw = sW + ((size_t)w * NK) * 32 * 16 + (size_t)lane * 16;
  #pragma unroll
  for (int ks = 0; ks < NK; ++ks) {
    v16h b = *(const v16h*)(sWw + (size_t)ks * 32 * 16);
    acc = __builtin_amdgcn_wmma_f32_16x16x32_f16(false, a[ks], false, b,
                                                 (short)0, acc, false, false);
  }

  const int ncol = (ntg0 + w) * 16 + r16;
  const int mbase = m0 + (lane >> 4) * 8;
  #pragma unroll
  for (int j = 0; j < 8; ++j) {
    int m = mbase + j;
    if (elem[m] == el) D[(size_t)m * ND + ncol] = acc[j];
  }
}

// ---------------- energy readout: out[g] += aew[elem] + h2 . w_read ----------------
__global__ void k_energy(const int* __restrict__ elem, const int* __restrict__ batch,
                         const float* __restrict__ h2, const float* __restrict__ wread,
                         const float* __restrict__ aew, float* __restrict__ out) {
  int n = blockIdx.x * blockDim.x + threadIdx.x;
  if (n >= Nn) return;
  float s = aew[elem[n]];
  const float* hr = h2 + (size_t)n * Cc;
  for (int c = 0; c < Cc; ++c) s += hr[c] * wread[c];
  atomicAdd(&out[batch[n]], s);
}

// ---------------- dB2 per element (dh2 is rank-1 = w_read) ----------------
__global__ void k_dB2elem(const float* __restrict__ Wp1, const float* __restrict__ wread,
                          float* __restrict__ dB2e) {
  int el = blockIdx.x, f = threadIdx.x;
  const float* Wr = Wp1 + (size_t)el * Ff * Cc + (size_t)f * Cc;
  float s = 0.0f;
  for (int c = 0; c < Cc; ++c) s += Wr[c] * wread[c];
  dB2e[el * Ff + f] = s;
}

// ---------------- dA = 2*A*dB (per-element dB table, layer 2) ----------------
__global__ void k_dA_tab(const float* __restrict__ A, const float* __restrict__ dBe,
                         const int* __restrict__ elem, float* __restrict__ dA) {
  int t = blockIdx.x * blockDim.x + threadIdx.x;
  if (t >= Nn * LMc * Cc) return;
  int n = t >> 10, lm = (t >> 6) & 15, c = t & 63;
  dA[t] = 2.0f * A[t] * dBe[elem[n] * Ff + lblock(lm) * Cc + c];
}

// ---------------- dA = 2*A*dB (per-node dB, layer 1) ----------------
__global__ void k_dA_node(const float* __restrict__ A, const float* __restrict__ dB,
                          float* __restrict__ dA) {
  int t = blockIdx.x * blockDim.x + threadIdx.x;
  if (t >= Nn * LMc * Cc) return;
  int n = t >> 10, lm = (t >> 6) & 15, c = t & 63;
  dA[t] = 2.0f * A[t] * dB[(size_t)n * Ff + lblock(lm) * Cc + c];
}

// ---------------- backward through message: dY, dw, dh[snd] ----------------
__global__ void k_bwd_msg(const int* __restrict__ eidx, const float* __restrict__ wgt,
                          const float* __restrict__ hprev, const float* __restrict__ Y,
                          const float* __restrict__ dA, float* __restrict__ dw,
                          float* __restrict__ dYacc, float* __restrict__ dh) {
  __shared__ float sY[LMc];
  __shared__ float sred[LMc * Cc];
  int e = blockIdx.x, c = threadIdx.x;
  if (c < LMc) sY[c] = Y[(size_t)e * LMc + c];
  __syncthreads();
  int snd = eidx[e], rcv = eidx[Ne + e];
  float w = wgt[(size_t)e * Cc + c];
  float hv = hprev[(size_t)snd * Cc + c];
  float wh = w * hv;
  const float* dAr = dA + (size_t)rcv * (LMc * Cc) + c;
  float g[LMc]; float dwh = 0.0f;
  #pragma unroll
  for (int lm = 0; lm < LMc; ++lm) {
    g[lm] = dAr[lm * Cc];
    dwh += g[lm] * sY[lm];
    sred[lm * Cc + c] = g[lm] * wh;
  }
  dw[(size_t)e * Cc + c] = dwh * hv;
  if (dh) atomicAdd(&dh[(size_t)snd * Cc + c], dwh * w);
  __syncthreads();
  if (c < LMc) {
    float s = 0.0f;
    for (int j = 0; j < Cc; ++j) s += sred[c * Cc + j];
    dYacc[(size_t)e * LMc + c] += s;
  }
}

// ---------------- dR[e,b] += sum_c dw[e,c] * W_rad[b,c] ----------------
__global__ void k_dR(const float* __restrict__ dw, const float* __restrict__ Wr,
                     float* __restrict__ dRacc) {
  int t = blockIdx.x * blockDim.x + threadIdx.x;
  if (t >= Ne * NBas) return;
  int e = t >> 3, b = t & 7;
  const float* dwr = dw + (size_t)e * Cc;
  const float* Wb = Wr + b * Cc;
  float s = 0.0f;
  for (int c = 0; c < Cc; ++c) s += dwr[c] * Wb[c];
  dRacc[t] += s;
}

// ---------------- transpose W_prod layer 0: Wt[el,c,f] = Wp[el,f,c] ----------------
__global__ void k_transpose(const float* __restrict__ Wp, float* __restrict__ Wt) {
  int t = blockIdx.x * blockDim.x + threadIdx.x;
  if (t >= NELEM * Ff * Cc) return;
  int el = t / (Ff * Cc), r = t % (Ff * Cc);
  int f = r / Cc, c = r % Cc;
  Wt[(size_t)el * Cc * Ff + (size_t)c * Ff + f] = Wp[t];
}

// ---------------- geometry backward -> forces ----------------
__global__ void k_geo_bwd(const int* __restrict__ eidx, const float* __restrict__ uu,
                          const float* __restrict__ rr, const float* __restrict__ dYacc,
                          const float* __restrict__ dRacc, const float* __restrict__ dRdr,
                          float* __restrict__ F) {
  int e = blockIdx.x * blockDim.x + threadIdx.x;
  if (e >= Ne) return;
  float x = uu[e * 3 + 0], y = uu[e * 3 + 1], z = uu[e * 3 + 2];
  float r = rr[e], inv = 1.0f / r;
  float x2 = x * x, y2 = y * y, z2 = z * z;
  const float* d = dYacc + (size_t)e * LMc;
  const float c1 = 0.4886025119029199f, c4 = 1.0925484305920792f;
  const float c6 = 0.31539156525252005f, c8 = 0.5462742152960396f;
  const float c9 = 0.5900435899266435f, c10 = 2.890611442640554f;
  const float c11 = 0.4570457994644658f, c12 = 0.3731763325901154f;
  const float c14 = 1.445305721320277f;
  float gx = d[3] * c1 + d[4] * c4 * y + d[7] * c4 * z + d[8] * 2.0f * c8 * x
           + d[9] * c9 * 6.0f * x * y + d[10] * c10 * y * z
           + d[13] * c11 * (5.0f * z2 - 1.0f) + d[14] * c14 * 2.0f * x * z
           + d[15] * c9 * (3.0f * x2 - 3.0f * y2);
  float gy = d[1] * c1 + d[4] * c4 * x + d[5] * c4 * z - d[8] * 2.0f * c8 * y
           + d[9] * c9 * (3.0f * x2 - 3.0f * y2) + d[10] * c10 * x * z
           + d[11] * c11 * (5.0f * z2 - 1.0f) - d[14] * c14 * 2.0f * y * z
           - d[15] * c9 * 6.0f * x * y;
  float gz = d[2] * c1 + d[5] * c4 * y + d[6] * c6 * 6.0f * z + d[7] * c4 * x
           + d[10] * c10 * x * y + d[11] * c11 * 10.0f * y * z
           + d[12] * c12 * (15.0f * z2 - 3.0f) + d[13] * c11 * 10.0f * x * z
           + d[14] * c14 * (x2 - y2);
  float dr = 0.0f;
  for (int b = 0; b < NBas; ++b)
    dr += dRacc[(size_t)e * NBas + b] * dRdr[(size_t)e * NBas + b];
  float gdotu = gx * x + gy * y + gz * z;
  float dvx = (gx - gdotu * x) * inv + dr * x;
  float dvy = (gy - gdotu * y) * inv + dr * y;
  float dvz = (gz - gdotu * z) * inv + dr * z;
  int snd = eidx[e], rcv = eidx[Ne + e];
  // F = -grad; dE/dpos[rcv] = +dvec, dE/dpos[snd] = -dvec
  atomicAdd(&F[snd * 3 + 0],  dvx); atomicAdd(&F[snd * 3 + 1],  dvy);
  atomicAdd(&F[snd * 3 + 2],  dvz);
  atomicAdd(&F[rcv * 3 + 0], -dvx); atomicAdd(&F[rcv * 3 + 1], -dvy);
  atomicAdd(&F[rcv * 3 + 2], -dvz);
}

// ---------------- host launcher ----------------
extern "C" void kernel_launch(void* const* d_in, const int* in_sizes, int n_in,
                              void* d_out, int out_size, void* d_ws, size_t ws_size,
                              hipStream_t stream) {
  const float* positions  = (const float*)d_in[0];
  const float* node_attrs = (const float*)d_in[1];
  const float* shifts     = (const float*)d_in[2];
  const float* W_embed    = (const float*)d_in[3];
  const float* aew        = (const float*)d_in[4];
  const float* W_rad      = (const float*)d_in[5];
  const float* W_prod     = (const float*)d_in[6];
  const float* w_read     = (const float*)d_in[7];
  const int*   edge_index = (const int*)d_in[8];
  const int*   batch      = (const int*)d_in[9];
  float* out = (float*)d_out;
  float* F   = out + Gg;

  char* base = (char*)d_ws; size_t off = 0;
  auto alloc = [&](size_t nbytes) -> void* {
    off = (off + 255) & ~(size_t)255;
    void* r = base + off; off += nbytes; return r;
  };
  int*      elem  = (int*)     alloc((size_t)Nn * 4);
  float*    uu    = (float*)   alloc((size_t)Ne * 3 * 4);
  float*    rr    = (float*)   alloc((size_t)Ne * 4);
  float*    Y     = (float*)   alloc((size_t)Ne * LMc * 4);
  float*    Rb    = (float*)   alloc((size_t)Ne * NBas * 4);
  float*    dRdr  = (float*)   alloc((size_t)Ne * NBas * 4);
  float*    h0    = (float*)   alloc((size_t)Nn * Cc * 4);
  float*    h1    = (float*)   alloc((size_t)Nn * Cc * 4);
  float*    h2    = (float*)   alloc((size_t)Nn * Cc * 4);
  float*    wgt0  = (float*)   alloc((size_t)Ne * Cc * 4);
  float*    wgt1  = (float*)   alloc((size_t)Ne * Cc * 4);
  float*    A1    = (float*)   alloc((size_t)Nn * LMc * Cc * 4);
  float*    A2    = (float*)   alloc((size_t)Nn * LMc * Cc * 4);
  float*    B1    = (float*)   alloc((size_t)Nn * Ff * 4);
  float*    B2    = (float*)   alloc((size_t)Nn * Ff * 4);
  float*    dh1   = (float*)   alloc((size_t)Nn * Cc * 4);
  float*    dB1   = (float*)   alloc((size_t)Nn * Ff * 4);
  float*    dA    = (float*)   alloc((size_t)Nn * LMc * Cc * 4);
  float*    dYacc = (float*)   alloc((size_t)Ne * LMc * 4);
  float*    dRacc = (float*)   alloc((size_t)Ne * NBas * 4);
  float*    dw    = (float*)   alloc((size_t)Ne * Cc * 4);
  float*    Wt0   = (float*)   alloc((size_t)NELEM * Ff * Cc * 4);
  float*    dB2e  = (float*)   alloc((size_t)NELEM * Ff * 4);
  _Float16* Wh0   = (_Float16*)alloc((size_t)NELEM * Ff * Cc * 2);
  _Float16* Wh1   = (_Float16*)alloc((size_t)NELEM * Ff * Cc * 2);
  _Float16* Whb   = (_Float16*)alloc((size_t)NELEM * Ff * Cc * 2);
  _Float16* Bh    = (_Float16*)alloc((size_t)Nn * Ff * 2);
  _Float16* dh1h  = (_Float16*)alloc((size_t)Nn * Cc * 2);
  (void)ws_size; (void)in_sizes; (void)n_in; (void)out_size;

  // zero accumulators + output
  (void)hipMemsetAsync(out, 0, (size_t)(Gg + Nn * 3) * 4, stream);
  (void)hipMemsetAsync(A1, 0, (size_t)Nn * LMc * Cc * 4, stream);
  (void)hipMemsetAsync(A2, 0, (size_t)Nn * LMc * Cc * 4, stream);
  (void)hipMemsetAsync(dh1, 0, (size_t)Nn * Cc * 4, stream);
  (void)hipMemsetAsync(dYacc, 0, (size_t)Ne * LMc * 4, stream);
  (void)hipMemsetAsync(dRacc, 0, (size_t)Ne * NBas * 4, stream);

  const float* Wp0 = W_prod;                                  // [10,256,64]
  const float* Wp1 = W_prod + (size_t)NELEM * Ff * Cc;
  const float* Wr0 = W_rad;
  const float* Wr1 = W_rad + NBas * Cc;

  k_elem<<<dim3((Nn + 127) / 128), dim3(128), 0, stream>>>(node_attrs, elem);
  k_embed<<<dim3(Nn * Cc / 256), dim3(256), 0, stream>>>(elem, W_embed, h0);
  k_geom<<<dim3(Ne / 256), dim3(256), 0, stream>>>(positions, shifts, edge_index,
                                                   uu, rr, Y, Rb, dRdr);
  // pre-pack forward weights into WMMA fragment layout (f16)
  k_packW<Ff, Cc><<<dim3(40), dim3(256), 0, stream>>>(Wp0, Wh0);
  k_packW<Ff, Cc><<<dim3(40), dim3(256), 0, stream>>>(Wp1, Wh1);

  // ---- layer 1 forward ----
  k_radW<<<dim3(Ne * Cc / 256), dim3(256), 0, stream>>>(Rb, Wr0, wgt0);
  k_scatter<<<dim3(Ne), dim3(Cc), 0, stream>>>(edge_index, wgt0, h0, Y, A1);
  k_poolB<<<dim3(Nn * Cc / 256), dim3(256), 0, stream>>>(A1, B1);
  k_cvt<<<dim3(Nn * Ff / 256), dim3(256), 0, stream>>>(B1, Bh, Nn * Ff);
  k_gemm2<Ff, Cc><<<dim3(Nn / 16, 1, NELEM), dim3(128), 0, stream>>>(Bh, Wh0, elem, h1);
  // ---- layer 2 forward ----
  k_radW<<<dim3(Ne * Cc / 256), dim3(256), 0, stream>>>(Rb, Wr1, wgt1);
  k_scatter<<<dim3(Ne), dim3(Cc), 0, stream>>>(edge_index, wgt1, h1, Y, A2);
  k_poolB<<<dim3(Nn * Cc / 256), dim3(256), 0, stream>>>(A2, B2);
  k_cvt<<<dim3(Nn * Ff / 256), dim3(256), 0, stream>>>(B2, Bh, Nn * Ff);
  k_gemm2<Ff, Cc><<<dim3(Nn / 16, 1, NELEM), dim3(128), 0, stream>>>(Bh, Wh1, elem, h2);
  // ---- energy ----
  k_energy<<<dim3((Nn + 127) / 128), dim3(128), 0, stream>>>(elem, batch, h2,
                                                             w_read, aew, out);
  // ---- backward layer 2 ----
  k_dB2elem<<<dim3(NELEM), dim3(Ff), 0, stream>>>(Wp1, w_read, dB2e);
  k_dA_tab<<<dim3(Nn * LMc * Cc / 256), dim3(256), 0, stream>>>(A2, dB2e, elem, dA);
  k_bwd_msg<<<dim3(Ne), dim3(Cc), 0, stream>>>(edge_index, wgt1, h1, Y, dA,
                                               dw, dYacc, dh1);
  k_dR<<<dim3(Ne * NBas / 256), dim3(256), 0, stream>>>(dw, Wr1, dRacc);
  // ---- backward layer 1 ----
  k_transpose<<<dim3(NELEM * Ff * Cc / 256), dim3(256), 0, stream>>>(Wp0, Wt0);
  k_packW<Cc, Ff><<<dim3(40), dim3(256), 0, stream>>>(Wt0, Whb);
  k_cvt<<<dim3(Nn * Cc / 256), dim3(256), 0, stream>>>(dh1, dh1h, Nn * Cc);
  k_gemm2<Cc, Ff><<<dim3(Nn / 16, Ff / 64, NELEM), dim3(128), 0, stream>>>(dh1h, Whb,
                                                                           elem, dB1);
  k_dA_node<<<dim3(Nn * LMc * Cc / 256), dim3(256), 0, stream>>>(A1, dB1, dA);
  k_bwd_msg<<<dim3(Ne), dim3(Cc), 0, stream>>>(edge_index, wgt0, h0, Y, dA,
                                               dw, dYacc, (float*)nullptr);
  k_dR<<<dim3(Ne * NBas / 256), dim3(256), 0, stream>>>(dw, Wr0, dRacc);
  // ---- geometry backward -> forces ----
  k_geo_bwd<<<dim3(Ne / 256), dim3(256), 0, stream>>>(edge_index, uu, rr, dYacc,
                                                      dRacc, dRdr, F);
}
